// VAELSTM_65859028517108
// MI455X (gfx1250) — matrix operand order repaired and checked by reading, not compiled, not run on hardware
//
#include <hip/hip_runtime.h>
#include <math.h>

// ---------------------------------------------------------------------------
// CDNA5 (gfx1250) wave32 WMMA types & helpers
// ---------------------------------------------------------------------------
typedef __attribute__((ext_vector_type(16))) _Float16 v16h;
typedef __attribute__((ext_vector_type(8)))  _Float16 h8;
typedef __attribute__((ext_vector_type(8)))  float    v8f;

__device__ __forceinline__ v8f wmma_f32_16x16x32(v16h a, v16h b, v8f c) {
  // 8 args: (neg_a, A, neg_b, B, c_mod, C, reuse_a, reuse_b)
  return __builtin_amdgcn_wmma_f32_16x16x32_f16(
      false, a, false, b, (short)0, c, false, false);
}

__device__ __forceinline__ v16h combine(h8 lo, h8 hi) {
  return __builtin_shufflevector(lo, hi, 0, 1, 2, 3, 4, 5, 6, 7, 8, 9, 10, 11,
                                 12, 13, 14, 15);
}

// A-fragment (16x32 f16): lane(half,r) holds K {h*8+0..7} U {16+h*8+0..7}
// => two contiguous 16B chunks of row r.
__device__ __forceinline__ int a_kpat(int half, int v, int e) {
  return ((v < 4) ? (half * 8 + 2 * v) : (16 + half * 8 + 2 * (v - 4))) + e;
}
__device__ __forceinline__ v16h a_frag_row(const _Float16* __restrict__ row,
                                           int half, int k0) {
  h8 lo = *(const h8*)(row + k0 + half * 8);
  h8 hi = *(const h8*)(row + k0 + 16 + half * 8);
  return combine(lo, hi);
}
// B-fragment (32x16 f16): lane(half,n) holds K {h*8+0..7} U {16+h*8+0..7} of column n;
// for C=A@W^T (W row-major [N][K]) this is two contiguous 16B chunks of row n.
__device__ __forceinline__ v16h b_frag_row(const _Float16* __restrict__ row,
                                           int half, int k0) {
  h8 lo = *(const h8*)(row + k0 + half * 8);
  h8 hi = *(const h8*)(row + k0 + 16 + half * 8);
  return combine(lo, hi);
}

__device__ __forceinline__ float gelu_exact(float x) {
  return 0.5f * x * (1.f + erff(x * 0.70710678118654752f));
}
__device__ __forceinline__ float sigmoidf(float x) {
  return 1.f / (1.f + __expf(-x));
}

// ---------------------------------------------------------------------------
// Tensor Data Mover: 2D tile (rows x elems of f16) global -> LDS
// D# fields per CDNA5 ISA 8.3/8.4 (group0 128b, group1 256b)
// ---------------------------------------------------------------------------
#if defined(__AMDGCN__) && __has_builtin(__builtin_amdgcn_tensor_load_to_lds)
#define USE_TDM 1
typedef __attribute__((ext_vector_type(4))) unsigned tdm_v4u;
typedef __attribute__((ext_vector_type(8))) int tdm_v8i;
typedef __attribute__((ext_vector_type(4))) int tdm_v4i;

__device__ __forceinline__ void tdm_load_2d_f16(unsigned lds_byte_addr,
                                                const void* gptr,
                                                unsigned elems, unsigned rows,
                                                unsigned long long stride) {
  unsigned long long ga = (unsigned long long)(uintptr_t)gptr;
  tdm_v4u g0;
  g0[0] = 1u;  // count=1, is_restore=0, gather off
  g0[1] = lds_byte_addr;
  g0[2] = (unsigned)(ga & 0xFFFFFFFFu);
  g0[3] = (unsigned)((ga >> 32) & 0x01FFFFFFu) | (2u << 30);  // type=2 (image)
  tdm_v8i g1;
  g1[0] = (int)(1u << 16);  // wg_mask=0, data_size=1 (2 bytes)
  g1[1] = (int)((elems & 0xFFFFu) << 16);              // tensor_dim0[15:0]
  g1[2] = (int)((elems >> 16) & 0xFFFFu) |             // tensor_dim0[31:16]
          (int)((rows & 0xFFFFu) << 16);               // tensor_dim1[15:0]
  g1[3] = (int)((rows >> 16) & 0xFFFFu) |              // tensor_dim1[31:16]
          (int)((elems & 0xFFFFu) << 16);              // tile_dim0
  g1[4] = (int)(rows & 0xFFFFu);                       // tile_dim1 (tile_dim2=0)
  g1[5] = (int)(stride & 0xFFFFFFFFull);               // dim0_stride[31:0]
  g1[6] = (int)((stride >> 32) & 0xFFFFull);           // dim0_stride[47:32]
  g1[7] = 0;
  tdm_v4i gz = {0, 0, 0, 0};
#if defined(__clang_major__) && (__clang_major__ >= 23)
  tdm_v8i gz8 = {0, 0, 0, 0, 0, 0, 0, 0};
  __builtin_amdgcn_tensor_load_to_lds(g0, g1, gz, gz, gz8, 0);
#else
  __builtin_amdgcn_tensor_load_to_lds(g0, g1, gz, gz, 0);
#endif
  __builtin_amdgcn_s_wait_tensorcnt(0);
}
#endif

// ---------------------------------------------------------------------------
// Pack kernels (run once per launch; all weights/activations -> f16)
// ---------------------------------------------------------------------------
__global__ void f32_to_f16(const float* __restrict__ s,
                           _Float16* __restrict__ d, int n) {
  int i = blockIdx.x * blockDim.x + threadIdx.x;
  if (i < n) d[i] = (_Float16)s[i];
}

// conv weight: OIHW fp32 -> [Cout][Kp] f16 with k = (dy*3+dx)*Cin + cin
__global__ void pack_conv_w(const float* __restrict__ w,
                            _Float16* __restrict__ wp, int Cout, int Cin,
                            int Kp) {
  int i = blockIdx.x * blockDim.x + threadIdx.x;
  if (i >= Cout * Kp) return;
  int n = i / Kp, k = i % Kp;
  float v = 0.f;
  if (k < Cin * 9) {
    int off9 = k / Cin, cin = k % Cin;
    int dy = off9 / 3, dx = off9 % 3;
    v = w[((n * Cin + cin) * 3 + dy) * 3 + dx];
  }
  wp[i] = (_Float16)v;
}

// enc_fc_w [256][4096] with k_old = c*64 + pix  ->  k_new = pix*64 + c (NHWC)
__global__ void pack_fc_w(const float* __restrict__ w,
                          _Float16* __restrict__ wp) {
  int i = blockIdx.x * blockDim.x + threadIdx.x;
  if (i >= 256 * 4096) return;
  int n = i >> 12, knew = i & 4095;
  int pix = knew >> 6, c = knew & 63;
  wp[i] = (_Float16)w[n * 4096 + (c * 64 + pix)];
}

// ---------------------------------------------------------------------------
// Conv 3x3 pad=1, NHWC f16 activations, implicit-GEMM WMMA.
// Block = 8 waves x 16x16 output tiles sharing one 16-row weight tile in LDS
// (staged by the Tensor Data Mover). K fully unrolled at compile time.
// ---------------------------------------------------------------------------
template <int CIN, int COUT, int H, int W, int STRIDE>
__global__ __launch_bounds__(256) void conv3x3_nhwc_wmma(
    const _Float16* __restrict__ in, const _Float16* __restrict__ wp,
    const float* __restrict__ bias, _Float16* __restrict__ out, int Nimg) {
  constexpr int HO = (STRIDE == 2) ? H / 2 : H;
  constexpr int WO = (STRIDE == 2) ? W / 2 : W;
  constexpr int K = CIN * 9;
  constexpr int KP = (K + 31) & ~31;
  constexpr int NT = COUT / 16;

  __shared__ __align__(32) _Float16 sh_w[16 * KP];

  const int P = Nimg * HO * WO;
  const int mtiles = (P + 15) >> 4;
  const int mblocks = (mtiles + 7) >> 3;
  const int bm = blockIdx.x % mblocks;
  const int nt = blockIdx.x / mblocks;

  // ---- stage this block's 16 x KP weight tile into LDS ----
  const _Float16* wrow = wp + (size_t)nt * 16 * KP;
#ifdef USE_TDM
  if (threadIdx.x < 32)  // wave 0 drives the TDM, waits TENSORcnt
    tdm_load_2d_f16((unsigned)(uintptr_t)sh_w, wrow, KP, 16, KP);
#else
  for (int i = threadIdx.x; i < 16 * KP; i += 256) sh_w[i] = wrow[i];
#endif
  __syncthreads();

  const int lane = threadIdx.x & 31;
  const int wv = threadIdx.x >> 5;
  const int mt = bm * 8 + wv;
  if (mt >= mtiles) return;  // wave-uniform; no barriers after this

  const int half = lane >> 4, col = lane & 15;
  const int m = mt * 16 + (lane & 15);
  const int px = (m < P) ? m : P - 1;
  const int img = px / (HO * WO);
  const int rem = px % (HO * WO);
  const int oy = rem / WO, ox = rem % WO;

  v8f acc = {};
  if constexpr (CIN == 1) {
    // K = 9 (padded to 32): scalar im2col gather, single WMMA
    v16h a, b;
#pragma unroll
    for (int v = 0; v < 8; ++v)
#pragma unroll
      for (int e = 0; e < 2; ++e) {
        int k = a_kpat(half, v, e);
        float av = 0.f;
        if (k < 9 && m < P) {
          int dy = k / 3, dx = k % 3;
          int iy = oy * STRIDE + dy - 1, ix = ox * STRIDE + dx - 1;
          if (iy >= 0 && iy < H && ix >= 0 && ix < W)
            av = (float)in[(img * H + iy) * W + ix];
        }
        a[2 * v + e] = (_Float16)av;
      }
    b = b_frag_row(&sh_w[col * KP], half, 0);
    acc = wmma_f32_16x16x32(a, b, acc);
  } else {
    const h8 zero = (h8)(_Float16)0.f;
#pragma unroll
    for (int c = 0; c < K / 32; ++c) {
      const int k0 = c * 32;
      const int off9 = k0 / CIN;           // fixed (dy,dx) per chunk
      const int cin0 = k0 % CIN;
      const int dy = off9 / 3, dx = off9 % 3;
      const int iy = oy * STRIDE + dy - 1, ix = ox * STRIDE + dx - 1;
      const bool valid = (m < P) & (iy >= 0) & (iy < H) & (ix >= 0) & (ix < W);
      const _Float16* ap =
          in + ((size_t)(img * H + iy) * W + ix) * CIN + cin0;
      h8 alo = valid ? *(const h8*)(ap + half * 8) : zero;
      h8 ahi = valid ? *(const h8*)(ap + 16 + half * 8) : zero;
      v16h a = combine(alo, ahi);
      v16h b = b_frag_row(&sh_w[col * KP], half, k0);
      acc = wmma_f32_16x16x32(a, b, acc);
    }
  }

  const int cc = nt * 16 + col;
  const float bz = bias[cc];
#pragma unroll
  for (int i = 0; i < 8; ++i) {
    int row = mt * 16 + i + 8 * half;  // row == NHWC pixel index
    if (row < P)
      out[(size_t)row * COUT + cc] = (_Float16)gelu_exact(acc[i] + bz);
  }
}

// ---------------------------------------------------------------------------
// Dense f16 GEMM: out[M x N](f32) = A[M x K] @ W^T + bias   (W is [N x K])
// ---------------------------------------------------------------------------
__global__ __launch_bounds__(256) void gemm_f16_wmma(
    const _Float16* __restrict__ A, const _Float16* __restrict__ Wt,
    const float* __restrict__ bias, float* __restrict__ out, int M, int N,
    int K) {
  const int lane = threadIdx.x & 31;
  const int wave = blockIdx.x * (blockDim.x >> 5) + (threadIdx.x >> 5);
  const int mtiles = (M + 15) >> 4;
  const int ntiles = N >> 4;
  if (wave >= mtiles * ntiles) return;
  const int mt = wave / ntiles, nt = wave % ntiles;
  const int half = lane >> 4, col = lane & 15;
  const int m = mt * 16 + (lane & 15);
  const _Float16* arow = A + (size_t)((m < M) ? m : M - 1) * K;
  const _Float16* brow = Wt + (size_t)(nt * 16 + col) * K;

  v8f acc = {};
#pragma unroll 4
  for (int k0 = 0; k0 < K; k0 += 32) {
    __builtin_prefetch(brow + k0 + 256, 0, 1);
    v16h a = a_frag_row(arow, half, k0);
    v16h b = b_frag_row(brow, half, k0);
    acc = wmma_f32_16x16x32(a, b, acc);
  }
  const int cc = nt * 16 + col;
  const float bz = bias[cc];
#pragma unroll
  for (int i = 0; i < 8; ++i) {
    int row = mt * 16 + i + 8 * half;
    if (row < M) out[(size_t)row * N + cc] = acc[i] + bz;
  }
}

// ---------------------------------------------------------------------------
// z = mu + eps * exp(0.5*logvar); writes f32 (loss) and f16 (LSTM input)
// ---------------------------------------------------------------------------
__device__ __forceinline__ unsigned pcg_hash(unsigned x) {
  x = x * 747796405u + 2891336453u;
  unsigned w = ((x >> ((x >> 28) + 4u)) ^ x) * 277803737u;
  return (w >> 22) ^ w;
}
__global__ void sample_z(const float* __restrict__ fcout,
                         float* __restrict__ z32, _Float16* __restrict__ z16,
                         int M) {
  int idx = blockIdx.x * blockDim.x + threadIdx.x;
  if (idx >= M * 128) return;
  int m = idx >> 7, j = idx & 127;
  float mu = fcout[m * 256 + j];
  float lv = fcout[m * 256 + 128 + j];
  unsigned h1 = pcg_hash(idx * 2u + 0x9e3779b9u);
  unsigned h2 = pcg_hash(idx * 2u + 1u + 0x85ebca6bu);
  float u1 = ((float)(h1 >> 8) + 1.f) * (1.f / 16777216.f);
  float u2 = (float)(h2 >> 8) * (1.f / 16777216.f);
  float eps = sqrtf(-2.f * __logf(u1)) * __cosf(6.2831853071795865f * u2);
  float zz = mu + eps * __expf(0.5f * lv);
  z32[idx] = zz;
  z16[idx] = (_Float16)zz;
}

// ---------------------------------------------------------------------------
// Fused 5-layer LSTM (B=16, HID=128) + linear head. One 1024-thread WG =
// 32 wave32 waves; each wave owns a 16-wide gate tile. h-state f16 in LDS
// (in-place layer update gives h_t[l-1] / h_{t-1}[l] semantics), gates f32.
// ---------------------------------------------------------------------------
#define HID 128
__global__ __launch_bounds__(1024) void lstm_fused_wmma(
    const _Float16* __restrict__ z16,    // [16*65 x 128]
    const _Float16* __restrict__ wih,    // [5 x 512 x 128] f16
    const _Float16* __restrict__ whh,    // [5 x 512 x 128] f16
    const float* __restrict__ bih, const float* __restrict__ bhh,
    const _Float16* __restrict__ lw,     // [128 x 128] f16
    const float* __restrict__ lin_b,
    float* __restrict__ pred,            // [16 x 64 x 128]
    float* __restrict__ cstate) {        // [5 x 16 x 128]
  __shared__ __align__(32) float sh_g[16][512];        // 32 KB
  __shared__ __align__(32) _Float16 sh_h[5][16][HID];  // 20 KB
  __shared__ __align__(32) _Float16 sh_x0[16][HID];    //  4 KB

  const int tid = threadIdx.x;
  const int lane = tid & 31;
  const int wv = tid >> 5;
  const int half = lane >> 4, col = lane & 15, r = lane & 15;

  for (int i = tid; i < 5 * 16 * HID; i += 1024) {
    ((_Float16*)sh_h)[i] = (_Float16)0.f;
    cstate[i] = 0.f;
  }
  __syncthreads();

  for (int t = 0; t < 64; ++t) {
    for (int it = 0; it < 2; ++it) {  // stage x_t
      int idx = tid + it * 1024;
      int b = idx >> 7, j = idx & 127;
      sh_x0[b][j] = z16[(b * 65 + t) * HID + j];
    }
    __syncthreads();

    for (int l = 0; l < 5; ++l) {
      const _Float16* xin = (l == 0) ? &sh_x0[0][0] : &sh_h[l - 1][0][0];
      const _Float16* hin = &sh_h[l][0][0];
      const int n = wv * 16 + col;
      const _Float16* wi = wih + ((size_t)l * 512 + n) * HID;
      const _Float16* wh = whh + ((size_t)l * 512 + n) * HID;
      v8f acc = {};
#pragma unroll
      for (int ch = 0; ch < 4; ++ch) {  // x @ Wih^T
        v16h a = a_frag_row(xin + r * HID, half, ch * 32);
        v16h b = b_frag_row(wi, half, ch * 32);
        acc = wmma_f32_16x16x32(a, b, acc);
      }
#pragma unroll
      for (int ch = 0; ch < 4; ++ch) {  // h @ Whh^T
        v16h a = a_frag_row(hin + r * HID, half, ch * 32);
        v16h b = b_frag_row(wh, half, ch * 32);
        acc = wmma_f32_16x16x32(a, b, acc);
      }
      const float bz = bih[l * 512 + n] + bhh[l * 512 + n];
#pragma unroll
      for (int i = 0; i < 8; ++i) sh_g[i + 8 * half][n] = acc[i] + bz;
      __syncthreads();

      for (int it = 0; it < 2; ++it) {  // cell update
        int idx = tid + it * 1024;
        int b = idx >> 7, j = idx & 127;
        float gi = sh_g[b][j];
        float gf = sh_g[b][128 + j];
        float gg = sh_g[b][256 + j];
        float go = sh_g[b][384 + j];
        float cprev = cstate[(l * 16 + b) * HID + j];
        float cnew = sigmoidf(gf) * cprev + sigmoidf(gi) * tanhf(gg);
        float hnew = sigmoidf(go) * tanhf(cnew);
        cstate[(l * 16 + b) * HID + j] = cnew;
        sh_h[l][b][j] = (_Float16)hnew;
      }
      __syncthreads();
    }

    if (wv < 8) {  // linear head: pred_t = h5 @ lin_w^T + lin_b
      const int n = wv * 16 + col;
      const _Float16* br = lw + (size_t)n * HID;
      v8f acc = {};
#pragma unroll
      for (int ch = 0; ch < 4; ++ch) {
        v16h a = a_frag_row(&sh_h[4][r][0], half, ch * 32);
        v16h b = b_frag_row(br, half, ch * 32);
        acc = wmma_f32_16x16x32(a, b, acc);
      }
      const float bz = lin_b[n];
#pragma unroll
      for (int i = 0; i < 8; ++i)
        pred[((i + 8 * half) * 64 + t) * HID + n] = acc[i] + bz;
    }
    __syncthreads();
  }
}

// ---------------------------------------------------------------------------
// Loss = mean((pred[:,3:] - z[:,4:65])^2)
// ---------------------------------------------------------------------------
__global__ void zero1(float* p) { p[0] = 0.f; }

__global__ void mse_partial(const float* __restrict__ pred,
                            const float* __restrict__ z,
                            float* __restrict__ acc) {
  __shared__ float red[256];
  int idx = blockIdx.x * blockDim.x + threadIdx.x;
  float v = 0.f;
  if (idx < 16 * 61 * 128) {
    int j = idx & 127;
    int tmp = idx >> 7;
    int t = 3 + tmp % 61;
    int b = tmp / 61;
    float d = pred[(b * 64 + t) * 128 + j] - z[(b * 65 + t + 1) * 128 + j];
    v = d * d;
  }
  red[threadIdx.x] = v;
  __syncthreads();
  for (int s = 128; s > 0; s >>= 1) {
    if (threadIdx.x < s) red[threadIdx.x] += red[threadIdx.x + s];
    __syncthreads();
  }
  if (threadIdx.x == 0) atomicAdd(acc, red[0]);
}

__global__ void finalize_loss(const float* __restrict__ acc,
                              float* __restrict__ out) {
  out[0] = acc[0] / (float)(16 * 61 * 128);
}

// ---------------------------------------------------------------------------
// Host launcher
// ---------------------------------------------------------------------------
static inline int cdiv(int a, int b) { return (a + b - 1) / b; }

extern "C" void kernel_launch(void* const* d_in, const int* in_sizes, int n_in,
                              void* d_out, int out_size, void* d_ws,
                              size_t ws_size, hipStream_t stream) {
  (void)in_sizes; (void)n_in; (void)out_size; (void)ws_size;
  const float* x       = (const float*)d_in[0];
  const float* enc_w1  = (const float*)d_in[1];
  const float* enc_b1  = (const float*)d_in[2];
  const float* enc_w2  = (const float*)d_in[3];
  const float* enc_b2  = (const float*)d_in[4];
  const float* enc_w3  = (const float*)d_in[5];
  const float* enc_b3  = (const float*)d_in[6];
  const float* enc_w4  = (const float*)d_in[7];
  const float* enc_b4  = (const float*)d_in[8];
  const float* enc_w5  = (const float*)d_in[9];
  const float* enc_b5  = (const float*)d_in[10];
  const float* enc_fcw = (const float*)d_in[11];
  const float* enc_fcb = (const float*)d_in[12];
  const float* lstm_Wih = (const float*)d_in[25];
  const float* lstm_Whh = (const float*)d_in[26];
  const float* lstm_bih = (const float*)d_in[27];
  const float* lstm_bhh = (const float*)d_in[28];
  const float* lin_w    = (const float*)d_in[29];
  const float* lin_b    = (const float*)d_in[30];
  float* out = (float*)d_out;

  // ---- workspace carve-up (64B aligned) ----
  char* p = (char*)d_ws;
  auto alloc = [&](size_t bytes) {
    char* q = p;
    p += (bytes + 63) & ~(size_t)63;
    return q;
  };
  const size_t NIMG = 1040;
  _Float16* xf16 = (_Float16*)alloc(NIMG * 4096 * 2);
  _Float16* actA = (_Float16*)alloc(NIMG * 32 * 32 * 32 * 2);
  _Float16* actB = (_Float16*)alloc(NIMG * 32 * 32 * 32 * 2);
  _Float16* wp1  = (_Float16*)alloc(32 * 32 * 2);
  _Float16* wp2  = (_Float16*)alloc(32 * 288 * 2);
  _Float16* wp3  = (_Float16*)alloc(64 * 288 * 2);
  _Float16* wp4  = (_Float16*)alloc(64 * 576 * 2);
  _Float16* wp5  = (_Float16*)alloc(64 * 576 * 2);
  _Float16* wfc  = (_Float16*)alloc(256 * 4096 * 2);
  _Float16* wih16 = (_Float16*)alloc(5 * 512 * 128 * 2);
  _Float16* whh16 = (_Float16*)alloc(5 * 512 * 128 * 2);
  _Float16* lw16  = (_Float16*)alloc(128 * 128 * 2);
  float* fcout = (float*)alloc(NIMG * 256 * 4);
  float* zbuf  = (float*)alloc(NIMG * 128 * 4);
  _Float16* z16 = (_Float16*)alloc(NIMG * 128 * 2);
  float* pred  = (float*)alloc(16 * 64 * 128 * 4);
  float* cst   = (float*)alloc(5 * 16 * 128 * 4);
  float* acc   = (float*)alloc(64);

  // ---- pack to f16 ----
  f32_to_f16<<<cdiv(NIMG * 4096, 256), 256, 0, stream>>>(x, xf16,
                                                         NIMG * 4096);
  pack_conv_w<<<cdiv(32 * 32, 256), 256, 0, stream>>>(enc_w1, wp1, 32, 1, 32);
  pack_conv_w<<<cdiv(32 * 288, 256), 256, 0, stream>>>(enc_w2, wp2, 32, 32, 288);
  pack_conv_w<<<cdiv(64 * 288, 256), 256, 0, stream>>>(enc_w3, wp3, 64, 32, 288);
  pack_conv_w<<<cdiv(64 * 576, 256), 256, 0, stream>>>(enc_w4, wp4, 64, 64, 576);
  pack_conv_w<<<cdiv(64 * 576, 256), 256, 0, stream>>>(enc_w5, wp5, 64, 64, 576);
  pack_fc_w<<<cdiv(256 * 4096, 256), 256, 0, stream>>>(enc_fcw, wfc);
  f32_to_f16<<<cdiv(5 * 512 * 128, 256), 256, 0, stream>>>(lstm_Wih, wih16,
                                                           5 * 512 * 128);
  f32_to_f16<<<cdiv(5 * 512 * 128, 256), 256, 0, stream>>>(lstm_Whh, whh16,
                                                           5 * 512 * 128);
  f32_to_f16<<<cdiv(128 * 128, 256), 256, 0, stream>>>(lin_w, lw16, 128 * 128);

  // ---- encoder conv stack (NHWC f16, TDM-staged weights) ----
  auto grid = [&](int P, int Cout) {
    return cdiv(cdiv(P, 16), 8) * (Cout / 16);
  };
  conv3x3_nhwc_wmma<1, 32, 64, 64, 2>
      <<<grid(NIMG * 32 * 32, 32), 256, 0, stream>>>(xf16, wp1, enc_b1, actA,
                                                     NIMG);
  conv3x3_nhwc_wmma<32, 32, 32, 32, 1>
      <<<grid(NIMG * 32 * 32, 32), 256, 0, stream>>>(actA, wp2, enc_b2, actB,
                                                     NIMG);
  conv3x3_nhwc_wmma<32, 64, 32, 32, 2>
      <<<grid(NIMG * 16 * 16, 64), 256, 0, stream>>>(actB, wp3, enc_b3, actA,
                                                     NIMG);
  conv3x3_nhwc_wmma<64, 64, 16, 16, 1>
      <<<grid(NIMG * 16 * 16, 64), 256, 0, stream>>>(actA, wp4, enc_b4, actB,
                                                     NIMG);
  conv3x3_nhwc_wmma<64, 64, 16, 16, 2>
      <<<grid(NIMG * 8 * 8, 64), 256, 0, stream>>>(actB, wp5, enc_b5, actA,
                                                   NIMG);

  // ---- enc FC (A = conv5 NHWC f16, W remapped) ----
  {
    int waves = cdiv(1040, 16) * (256 / 16);
    gemm_f16_wmma<<<cdiv(waves, 8), 256, 0, stream>>>(actA, wfc, enc_fcb,
                                                      fcout, 1040, 256, 4096);
  }

  sample_z<<<cdiv(1040 * 128, 256), 256, 0, stream>>>(fcout, zbuf, z16, 1040);

  lstm_fused_wmma<<<1, 1024, 0, stream>>>(z16, wih16, whh16, lstm_bih,
                                          lstm_bhh, lw16, lin_b, pred, cst);

  zero1<<<1, 1, 0, stream>>>(acc);
  mse_partial<<<cdiv(16 * 61 * 128, 256), 256, 0, stream>>>(pred, zbuf, acc);
  finalize_loss<<<1, 1, 0, stream>>>(acc, out);
}
